// Rwkv7SelfAttention_20100446945849
// MI455X (gfx1250) — compile-verified, matrix operand order checked by reading, not run on hardware
//
#include <hip/hip_runtime.h>
#include <cstdint>
#include <cstddef>

#define S_LEN 1024
#define HID   2048
#define NH    32
#define HS    64
#define DD    96
#define DA    96
#define DV    64
#define DG    256
#define SH    (S_LEN*HID)

typedef __bf16 bf16_t;
typedef bf16_t v16bf __attribute__((ext_vector_type(16)));
typedef bf16_t v8bf  __attribute__((ext_vector_type(8)));
typedef float  v8f   __attribute__((ext_vector_type(8)));

__device__ __forceinline__ float sigmoidf(float x){ return 1.0f/(1.0f + expf(-x)); }

__device__ __forceinline__ v16bf load_frag16(const bf16_t* p0, const bf16_t* p1){
  v8bf lo = *(const v8bf*)p0;
  v8bf hi = *(const v8bf*)p1;
  return __builtin_shufflevector(lo, hi, 0,1,2,3,4,5,6,7,8,9,10,11,12,13,14,15);
}

__device__ __forceinline__ void store_tile(float* __restrict__ C, int N,
                                           int m0, int n0, int h, int l, v8f acc){
  float* cp = C + (size_t)(m0 + h*8)*N + n0 + l;
  #pragma unroll
  for (int j = 0; j < 8; ++j) cp[(size_t)j*N] = acc[j];
}

// ---------------------------------------------------------------------------
// Weight convert + transpose: W [K][N] f32 -> WT [N][K] bf16
// ---------------------------------------------------------------------------
__global__ void k_transpose_bf16(const float* __restrict__ W, bf16_t* __restrict__ WT,
                                 int K, int N){
  size_t id = (size_t)blockIdx.x*256 + threadIdx.x;
  if (id >= (size_t)K*N) return;
  int kk = (int)(id / N), n = (int)(id % N);
  WT[(size_t)n*K + kk] = (bf16_t)W[id];
}

// ---------------------------------------------------------------------------
// LayerNorm over HID per token
// ---------------------------------------------------------------------------
__global__ void k_layernorm(const float* __restrict__ x, const float* __restrict__ w,
                            const float* __restrict__ b, float* __restrict__ xn){
  const int s = blockIdx.x, t = threadIdx.x;
  const float* xi = x + (size_t)s*HID;
  float vals[8]; float sum = 0.f, sq = 0.f;
  #pragma unroll
  for (int j = 0; j < 8; ++j){ float v = xi[t + j*256]; vals[j]=v; sum += v; sq += v*v; }
  __shared__ float r1[256], r2[256];
  r1[t]=sum; r2[t]=sq; __syncthreads();
  for (int off = 128; off > 0; off >>= 1){
    if (t < off){ r1[t]+=r1[t+off]; r2[t]+=r2[t+off]; } __syncthreads();
  }
  float mean = r1[0]*(1.0f/HID);
  float var  = r2[0]*(1.0f/HID) - mean*mean;
  float inv  = rsqrtf(var + 1e-5f);
  float* xo = xn + (size_t)s*HID;
  #pragma unroll
  for (int j = 0; j < 8; ++j){ int i = t + j*256; xo[i] = (vals[j]-mean)*inv*w[i] + b[i]; }
}

// ---------------------------------------------------------------------------
// Token shift -> six bf16 mixed activations
// ---------------------------------------------------------------------------
__global__ void k_tokenshift(const float* __restrict__ xn, const float* __restrict__ state1,
                             const float* __restrict__ mr, const float* __restrict__ mw,
                             const float* __restrict__ mk, const float* __restrict__ mv,
                             const float* __restrict__ ma, const float* __restrict__ mg,
                             bf16_t* xr, bf16_t* xw, bf16_t* xk,
                             bf16_t* xv, bf16_t* xa, bf16_t* xg){
  size_t id = (size_t)blockIdx.x*256 + threadIdx.x;
  int s = (int)(id / HID), i = (int)(id % HID);
  float cur  = xn[id];
  float prev = (s == 0) ? state1[i] : xn[id - HID];
  float sx = prev - cur;
  xr[id] = (bf16_t)(cur + mr[i]*sx);
  xw[id] = (bf16_t)(cur + mw[i]*sx);
  xk[id] = (bf16_t)(cur + mk[i]*sx);
  xv[id] = (bf16_t)(cur + mv[i]*sx);
  xa[id] = (bf16_t)(cur + ma[i]*sx);
  xg[id] = (bf16_t)(cur + mg[i]*sx);
}

// ---------------------------------------------------------------------------
// WMMA bf16 GEMM, register-blocked: each wave computes a 32x32 C tile
// (2 A-frags x 2 B-frags -> 4 independent WMMAs per K-step).
// Block = 128 threads = 4 waves arranged 2x2 -> 64x64 C per block.
// grid = (N/64, M/64). Requires N%64==0, K%32==0, M%64==0.
// ---------------------------------------------------------------------------
__global__ void k_gemm_bf16_x4(const bf16_t* __restrict__ A, const bf16_t* __restrict__ BT,
                               float* __restrict__ C, int M, int N, int K){
  const int wave = threadIdx.x >> 5;
  const int lane = threadIdx.x & 31;
  const int h = lane >> 4, l = lane & 15;
  const int wm = wave >> 1, wn = wave & 1;
  const int m0 = (blockIdx.y << 6) + wm*32;
  const int n0 = (blockIdx.x << 6) + wn*32;
  const bf16_t* arow0 = A  + (size_t)(m0 + l)*K + h*8;
  const bf16_t* arow1 = arow0 + (size_t)16*K;
  const bf16_t* bcol0 = BT + (size_t)(n0 + l)*K + h*8;
  const bf16_t* bcol1 = bcol0 + (size_t)16*K;
  v8f acc00 = {}, acc01 = {}, acc10 = {}, acc11 = {};
  for (int kb = 0; kb < K; kb += 32){
    v16bf a0 = load_frag16(arow0 + kb, arow0 + kb + 16);
    v16bf a1 = load_frag16(arow1 + kb, arow1 + kb + 16);
    v16bf b0 = load_frag16(bcol0 + kb, bcol0 + kb + 16);
    v16bf b1 = load_frag16(bcol1 + kb, bcol1 + kb + 16);
    acc00 = __builtin_amdgcn_wmma_f32_16x16x32_bf16(false, a0, false, b0, (short)0, acc00, false, false);
    acc01 = __builtin_amdgcn_wmma_f32_16x16x32_bf16(false, a0, false, b1, (short)0, acc01, false, false);
    acc10 = __builtin_amdgcn_wmma_f32_16x16x32_bf16(false, a1, false, b0, (short)0, acc10, false, false);
    acc11 = __builtin_amdgcn_wmma_f32_16x16x32_bf16(false, a1, false, b1, (short)0, acc11, false, false);
  }
  store_tile(C, N, m0,      n0,      h, l, acc00);
  store_tile(C, N, m0,      n0 + 16, h, l, acc01);
  store_tile(C, N, m0 + 16, n0,      h, l, acc10);
  store_tile(C, N, m0 + 16, n0 + 16, h, l, acc11);
}

// ---------------------------------------------------------------------------
// WMMA bf16 GEMM, 1 tile/wave (fallback for N%64!=0, e.g. N=96)
// grid = (N/16, M/64), block = 128 (4 waves stacked in M)
// ---------------------------------------------------------------------------
__global__ void k_gemm_bf16(const bf16_t* __restrict__ A, const bf16_t* __restrict__ BT,
                            float* __restrict__ C, int M, int N, int K){
  const int wave = threadIdx.x >> 5;
  const int lane = threadIdx.x & 31;
  const int h = lane >> 4, l = lane & 15;
  const int n0 = blockIdx.x << 4;
  const int m0 = (blockIdx.y << 6) + (wave << 4);
  const bf16_t* arow = A  + (size_t)(m0 + l)*K + h*8;
  const bf16_t* bcol = BT + (size_t)(n0 + l)*K + h*8;
  v8f acc = {};
  for (int kb = 0; kb < K; kb += 32){
    v16bf af = load_frag16(arow + kb, arow + kb + 16);
    v16bf bfv = load_frag16(bcol + kb, bcol + kb + 16);
    acc = __builtin_amdgcn_wmma_f32_16x16x32_bf16(false, af, false, bfv,
                                                  (short)0, acc, false, false);
  }
  store_tile(C, N, m0, n0, h, l, acc);
}

// ---------------------------------------------------------------------------
// f32 -> bf16 with optional activation. mode: 0=copy, 1=tanh, 2=sigmoid
// ---------------------------------------------------------------------------
__global__ void k_act_bf16(const float* __restrict__ src, bf16_t* __restrict__ dst,
                           int n, int mode){
  int id = blockIdx.x*256 + threadIdx.x;
  if (id >= n) return;
  float v = src[id];
  if (mode == 1) v = tanhf(v);
  else if (mode == 2) v = sigmoidf(v);
  dst[id] = (bf16_t)v;
}

// ---------------------------------------------------------------------------
// Pre-scan elementwise: decay w, a-sigmoid, kk-normalize, k/v updates, A/B
// one block per (token, head), 64 threads
// ---------------------------------------------------------------------------
__global__ void k_prescan(float* __restrict__ wlin, float* __restrict__ kbuf,
                          float* __restrict__ vbuf,
                          const float* __restrict__ aaf, const float* __restrict__ vvf,
                          const float* __restrict__ v_first,
                          const float* __restrict__ w0, const float* __restrict__ a0,
                          const float* __restrict__ v0,
                          const float* __restrict__ k_k, const float* __restrict__ k_a,
                          float* __restrict__ Ab, float* __restrict__ Bb){
  const int blk = blockIdx.x;            // s*NH + head
  const int hd = blk % NH;
  const int t  = threadIdx.x;
  const int i  = hd*HS + t;              // index within HID
  const size_t gid = (size_t)blk*HS + t; // == s*HID + i
  float kf = kbuf[gid];
  float kk = kf * k_k[i];
  __shared__ float red[HS];
  red[t] = kk*kk; __syncthreads();
  for (int off = 32; off > 0; off >>= 1){
    if (t < off) red[t] += red[t+off];
    __syncthreads();
  }
  float nrm = fmaxf(sqrtf(red[0]), 1e-12f);
  float kkn = kk / nrm;
  float a_s = sigmoidf(a0[i] + aaf[gid]);
  float wd  = expf(-0.606531f * sigmoidf(w0[i] + wlin[gid]));
  float kn  = kf * (1.0f + (a_s - 1.0f) * k_a[i]);
  float vf  = vbuf[gid];
  float vn  = vf + (v_first[gid] - vf) * sigmoidf(v0[i] + vvf[gid]);
  wlin[gid] = wd;       // in-place: becomes decay w
  kbuf[gid] = kn;       // in-place: updated k
  vbuf[gid] = vn;       // in-place: updated v
  Ab[gid] = kkn * a_s;
  Bb[gid] = -kkn;
}

// ---------------------------------------------------------------------------
// Sequential scan: one block (64 threads) per head; 64x64 state in VGPRs
// (lane t owns column m=t); per-step 64-wide operands staged via LDS.
// Prefetch next step's operands to shorten the serial critical path.
// ---------------------------------------------------------------------------
__global__ void k_scan(const float* __restrict__ w, const float* __restrict__ r,
                       const float* __restrict__ k, const float* __restrict__ v,
                       const float* __restrict__ A, const float* __restrict__ B,
                       const float* __restrict__ s2_in,
                       float* __restrict__ y, float* __restrict__ s2_out){
  const int hd = blockIdx.x;
  const int t  = threadIdx.x;     // column m
  float s[HS];
  #pragma unroll
  for (int n = 0; n < HS; ++n) s[n] = s2_in[((size_t)hd*HS + n)*HS + t];
  __shared__ float ws[HS], rs[HS], ks[HS], as_[HS], bs_[HS];
  for (int step = 0; step < S_LEN; ++step){
    const size_t base = (size_t)step*HID + hd*HS;
    ws[t]  = w[base+t]; rs[t] = r[base+t]; ks[t] = k[base+t];
    as_[t] = A[base+t]; bs_[t] = B[base+t];
    float vm = v[base+t];
    if (step + 1 < S_LEN){
      const size_t nb = base + HID;
      __builtin_prefetch(&w[nb+t], 0, 1); __builtin_prefetch(&r[nb+t], 0, 1);
      __builtin_prefetch(&k[nb+t], 0, 1); __builtin_prefetch(&v[nb+t], 0, 1);
      __builtin_prefetch(&A[nb+t], 0, 1); __builtin_prefetch(&B[nb+t], 0, 1);
    }
    __syncthreads();
    float bsum = 0.f;
    #pragma unroll 16
    for (int n = 0; n < HS; ++n) bsum += bs_[n]*s[n];
    float ysum = 0.f;
    #pragma unroll 16
    for (int n = 0; n < HS; ++n){
      s[n] = s[n]*ws[n] + as_[n]*bsum + ks[n]*vm;
      ysum += rs[n]*s[n];
    }
    y[base+t] = ysum;
    __syncthreads();
  }
  #pragma unroll
  for (int n = 0; n < HS; ++n) s2_out[((size_t)hd*HS + n)*HS + t] = s[n];
}

// ---------------------------------------------------------------------------
// Post-scan: per-head groupnorm(eps=6.4e-4), lnx affine, bonus term, gate -> bf16
// one block per (token, head), 64 threads
// ---------------------------------------------------------------------------
__global__ void k_postscan(const float* __restrict__ y, const float* __restrict__ r,
                           const float* __restrict__ k, const float* __restrict__ v,
                           const float* __restrict__ gate,
                           const float* __restrict__ r_k,
                           const float* __restrict__ lnx_w, const float* __restrict__ lnx_b,
                           bf16_t* __restrict__ ygate){
  const int blk = blockIdx.x;
  const int hd = blk % NH;
  const int t  = threadIdx.x;
  const int i  = hd*HS + t;
  const size_t gid = (size_t)blk*HS + t;
  float yv = y[gid];
  __shared__ float r1[HS], r2[HS];
  r1[t] = yv; r2[t] = yv*yv; __syncthreads();
  for (int off = 32; off > 0; off >>= 1){
    if (t < off){ r1[t] += r1[t+off]; r2[t] += r2[t+off]; } __syncthreads();
  }
  float mean = r1[0]*(1.0f/HS);
  float var  = r2[0]*(1.0f/HS) - mean*mean;
  float yn = (yv - mean) * rsqrtf(var + 0.00064f);
  float y2 = yn * lnx_w[i] + lnx_b[i];
  __syncthreads();
  r1[t] = r[gid]*k[gid]*r_k[i]; __syncthreads();
  for (int off = 32; off > 0; off >>= 1){
    if (t < off) r1[t] += r1[t+off];
    __syncthreads();
  }
  y2 += r1[0] * v[gid];
  y2 *= gate[gid];
  ygate[gid] = (bf16_t)y2;
}

// ---------------------------------------------------------------------------
// Final residual add and output copies
// ---------------------------------------------------------------------------
__global__ void k_residual(const float* __restrict__ x, const float* __restrict__ yw,
                           float* __restrict__ out){
  size_t id = (size_t)blockIdx.x*256 + threadIdx.x;
  out[id] = x[id] + yw[id];
}
__global__ void k_copy_f32(const float* __restrict__ src, float* __restrict__ dst, int n){
  int id = blockIdx.x*256 + threadIdx.x;
  if (id < n) dst[id] = src[id];
}
__global__ void k_state1(const float* __restrict__ xn, float* __restrict__ s1){
  int i = blockIdx.x*256 + threadIdx.x;
  s1[i] = xn[(size_t)(S_LEN-1)*HID + i];
}

// ---------------------------------------------------------------------------
extern "C" void kernel_launch(void* const* d_in, const int* in_sizes, int n_in,
                              void* d_out, int out_size, void* d_ws, size_t ws_size,
                              hipStream_t stream){
  const float* x      = (const float*)d_in[0];
  const float* state1 = (const float*)d_in[1];
  const float* state2 = (const float*)d_in[2];
  const float* vfirst = (const float*)d_in[3];
  const float* x_r = (const float*)d_in[4];
  const float* x_w = (const float*)d_in[5];
  const float* x_k = (const float*)d_in[6];
  const float* x_v = (const float*)d_in[7];
  const float* x_a = (const float*)d_in[8];
  const float* x_g = (const float*)d_in[9];
  const float* ln1_w = (const float*)d_in[10];
  const float* ln1_b = (const float*)d_in[11];
  const float* w0 = (const float*)d_in[12];
  const float* a0 = (const float*)d_in[13];
  const float* v0 = (const float*)d_in[14];
  const float* Wr = (const float*)d_in[15];
  const float* Wk = (const float*)d_in[16];
  const float* Wv = (const float*)d_in[17];
  const float* Wo = (const float*)d_in[18];
  const float* W1 = (const float*)d_in[19];
  const float* W2 = (const float*)d_in[20];
  const float* A1 = (const float*)d_in[21];
  const float* A2 = (const float*)d_in[22];
  const float* V1 = (const float*)d_in[23];
  const float* V2 = (const float*)d_in[24];
  const float* G1 = (const float*)d_in[25];
  const float* G2 = (const float*)d_in[26];
  const float* k_k = (const float*)d_in[27];
  const float* k_a = (const float*)d_in[28];
  const float* r_k = (const float*)d_in[29];
  const float* lnx_w = (const float*)d_in[30];
  const float* lnx_b = (const float*)d_in[31];

  // ---- carve workspace ----
  char* base = (char*)d_ws;
  size_t off = 0;
  auto carve = [&](size_t bytes)->char*{
    char* p = base + off; off += (bytes + 255) & ~(size_t)255; return p;
  };
  // bf16 buffers
  bf16_t* xr_b = (bf16_t*)carve((size_t)SH*2);
  bf16_t* xw_b = (bf16_t*)carve((size_t)SH*2);
  bf16_t* xk_b = (bf16_t*)carve((size_t)SH*2);
  bf16_t* xv_b = (bf16_t*)carve((size_t)SH*2);
  bf16_t* xa_b = (bf16_t*)carve((size_t)SH*2);
  bf16_t* xg_b = (bf16_t*)carve((size_t)SH*2);
  bf16_t* WrT = (bf16_t*)carve((size_t)HID*HID*2);
  bf16_t* WkT = (bf16_t*)carve((size_t)HID*HID*2);
  bf16_t* WvT = (bf16_t*)carve((size_t)HID*HID*2);
  bf16_t* WoT = (bf16_t*)carve((size_t)HID*HID*2);
  bf16_t* W1T = (bf16_t*)carve((size_t)HID*DD*2);
  bf16_t* W2T = (bf16_t*)carve((size_t)HID*DD*2);
  bf16_t* A1T = (bf16_t*)carve((size_t)HID*DA*2);
  bf16_t* A2T = (bf16_t*)carve((size_t)HID*DA*2);
  bf16_t* V1T = (bf16_t*)carve((size_t)HID*DV*2);
  bf16_t* V2T = (bf16_t*)carve((size_t)HID*DV*2);
  bf16_t* G1T = (bf16_t*)carve((size_t)HID*DG*2);
  bf16_t* G2T = (bf16_t*)carve((size_t)HID*DG*2);
  bf16_t* hwb = (bf16_t*)carve((size_t)S_LEN*DD*2);
  bf16_t* hab = (bf16_t*)carve((size_t)S_LEN*DA*2);
  bf16_t* hvb = (bf16_t*)carve((size_t)S_LEN*DV*2);
  bf16_t* hgb = (bf16_t*)carve((size_t)S_LEN*DG*2);
  bf16_t* ygb = (bf16_t*)carve((size_t)SH*2);
  // f32 buffers
  float* xn   = (float*)carve((size_t)SH*4);
  float* rbuf = (float*)carve((size_t)SH*4);
  float* kbuf = (float*)carve((size_t)SH*4);
  float* vbuf = (float*)carve((size_t)SH*4);
  float* hwf  = (float*)carve((size_t)S_LEN*DD*4);
  float* haf  = (float*)carve((size_t)S_LEN*DA*4);
  float* hvf  = (float*)carve((size_t)S_LEN*DV*4);
  float* hgf  = (float*)carve((size_t)S_LEN*DG*4);
  float* wlin = (float*)carve((size_t)SH*4);   // becomes decay w in-place
  float* aaf  = (float*)carve((size_t)SH*4);
  float* vvf  = (float*)carve((size_t)SH*4);
  float* gatef= (float*)carve((size_t)SH*4);
  float* Ab   = (float*)carve((size_t)SH*4);
  float* Bb   = (float*)carve((size_t)SH*4);
  float* ybuf = (float*)carve((size_t)SH*4);   // scan y, reused for y@Wo

  float* out_f = (float*)d_out;
  float* out_main = out_f;                       // (1,S,HID)
  float* out_s1   = out_f + (size_t)SH;          // (1,HID)
  float* out_s2   = out_s1 + HID;                // (NH,HS,HS)
  float* out_vf   = out_s2 + (size_t)NH*HS*HS;   // (1,S,HID)

  // ---- weight transposes (f32 -> bf16 [N][K]) ----
  auto T = [&](const float* W, bf16_t* WT, int K, int N){
    int total = K*N;
    k_transpose_bf16<<<dim3((total+255)/256), dim3(256), 0, stream>>>(W, WT, K, N);
  };
  T(Wr, WrT, HID, HID); T(Wk, WkT, HID, HID); T(Wv, WvT, HID, HID); T(Wo, WoT, HID, HID);
  T(W1, W1T, HID, DD);  T(W2, W2T, DD, HID);
  T(A1, A1T, HID, DA);  T(A2, A2T, DA, HID);
  T(V1, V1T, HID, DV);  T(V2, V2T, DV, HID);
  T(G1, G1T, HID, DG);  T(G2, G2T, DG, HID);

  // ---- LN + token shift ----
  k_layernorm<<<dim3(S_LEN), dim3(256), 0, stream>>>(x, ln1_w, ln1_b, xn);
  k_tokenshift<<<dim3(SH/256), dim3(256), 0, stream>>>(xn, state1,
      x_r, x_w, x_k, x_v, x_a, x_g, xr_b, xw_b, xk_b, xv_b, xa_b, xg_b);

  // ---- WMMA GEMMs (register-blocked when N%64==0) ----
  auto G = [&](const bf16_t* A_, const bf16_t* BT_, float* C_, int N, int K){
    if ((N & 63) == 0)
      k_gemm_bf16_x4<<<dim3(N/64, S_LEN/64), dim3(128), 0, stream>>>(A_, BT_, C_, S_LEN, N, K);
    else
      k_gemm_bf16<<<dim3(N/16, S_LEN/64), dim3(128), 0, stream>>>(A_, BT_, C_, S_LEN, N, K);
  };
  G(xr_b, WrT, rbuf, HID, HID);
  G(xk_b, WkT, kbuf, HID, HID);
  G(xv_b, WvT, vbuf, HID, HID);
  G(xw_b, W1T, hwf, DD, HID);
  G(xa_b, A1T, haf, DA, HID);
  G(xv_b, V1T, hvf, DV, HID);
  G(xg_b, G1T, hgf, DG, HID);

  // ---- hidden activations -> bf16 ----
  k_act_bf16<<<dim3((S_LEN*DD+255)/256), dim3(256), 0, stream>>>(hwf, hwb, S_LEN*DD, 1);
  k_act_bf16<<<dim3((S_LEN*DA+255)/256), dim3(256), 0, stream>>>(haf, hab, S_LEN*DA, 0);
  k_act_bf16<<<dim3((S_LEN*DV+255)/256), dim3(256), 0, stream>>>(hvf, hvb, S_LEN*DV, 0);
  k_act_bf16<<<dim3((S_LEN*DG+255)/256), dim3(256), 0, stream>>>(hgf, hgb, S_LEN*DG, 2);

  // ---- second-stage GEMMs ----
  G(hwb, W2T, wlin, HID, DD);
  G(hab, A2T, aaf,  HID, DA);
  G(hvb, V2T, vvf,  HID, DV);
  G(hgb, G2T, gatef, HID, DG);

  // ---- pre-scan elementwise ----
  k_prescan<<<dim3(S_LEN*NH), dim3(HS), 0, stream>>>(wlin, kbuf, vbuf, aaf, vvf,
      vfirst, w0, a0, v0, k_k, k_a, Ab, Bb);

  // ---- sequential scan ----
  k_scan<<<dim3(NH), dim3(HS), 0, stream>>>(wlin, rbuf, kbuf, vbuf, Ab, Bb,
      state2, ybuf, out_s2);

  // ---- post-scan (groupnorm + bonus + gate) ----
  k_postscan<<<dim3(S_LEN*NH), dim3(HS), 0, stream>>>(ybuf, rbuf, kbuf, vbuf,
      gatef, r_k, lnx_w, lnx_b, ygb);

  // ---- output projection + residual ----
  G(ygb, WoT, ybuf, HID, HID);   // ybuf reused as y@Wo
  k_residual<<<dim3(SH/256), dim3(256), 0, stream>>>(x, ybuf, out_main);

  // ---- remaining outputs ----
  k_state1<<<dim3(HID/256), dim3(256), 0, stream>>>(xn, out_s1);
  k_copy_f32<<<dim3(SH/256), dim3(256), 0, stream>>>(vfirst, out_vf, SH);

  (void)in_sizes; (void)n_in; (void)out_size; (void)ws_size;
}
